// GraphsageMEAN_64561948393902
// MI455X (gfx1250) — compile-verified, matrix operand order checked by reading, not run on hardware
//
#include <hip/hip_runtime.h>
#include <hip/hip_bf16.h>

// ---------------- bf16 <-> f32 helpers (bit-level, RNE) ----------------
typedef __attribute__((ext_vector_type(16))) __bf16        v16bf;
typedef __attribute__((ext_vector_type(8)))  float         v8f;
typedef __attribute__((ext_vector_type(8)))  unsigned short ushort8;

__device__ __forceinline__ unsigned short f2bf(float f) {
    unsigned int u = __float_as_uint(f);
    u += 0x7FFFu + ((u >> 16) & 1u);           // round-to-nearest-even
    return (unsigned short)(u >> 16);
}
__device__ __forceinline__ float bf2f(unsigned short h) {
    return __uint_as_float(((unsigned int)h) << 16);
}

// ---------------- elementwise f32 -> bf16 ----------------
__global__ void cvt_bf16_kernel(const float* __restrict__ in,
                                unsigned short* __restrict__ out, int total) {
    int i = blockIdx.x * blockDim.x + threadIdx.x;
    if (i < total) out[i] = f2bf(in[i]);
}

// ---------------- W[K][N] f32 -> Wt[N][K] bf16 (N-major for B fragments) ----
__global__ void wt_kernel(const float* __restrict__ W,
                          unsigned short* __restrict__ Wt, int K, int Nw) {
    int idx = blockIdx.x * blockDim.x + threadIdx.x;
    if (idx >= K * Nw) return;
    int k = idx / Nw, n = idx % Nw;
    Wt[(size_t)n * K + k] = f2bf(W[idx]);
}

// ---------------- in-degree counts ----------------
__global__ void count_kernel(const int* __restrict__ dst, float* __restrict__ cnt,
                             int nE) {
    int e = blockIdx.x * blockDim.x + threadIdx.x;
    if (e < nE) atomicAdd(cnt + dst[e], 1.0f);
}

// ---------------- edge scatter: agg[dst] += feat_bf16[src] (fp32 accum) ----
// one thread handles 4 contiguous features of one edge (8B bf16 load)
__global__ void scatter_kernel(const unsigned short* __restrict__ feat,
                               const int* __restrict__ src,
                               const int* __restrict__ dst,
                               float* __restrict__ agg,
                               int nE, int d, int log2cpe) {
    unsigned int gid = blockIdx.x * blockDim.x + threadIdx.x;
    unsigned int e = gid >> log2cpe;
    if (e >= (unsigned int)nE) return;
    unsigned int c = (gid & ((1u << log2cpe) - 1u)) << 2;   // feature offset
    int s = src[e], t = dst[e];
    const unsigned short* p = feat + (size_t)s * d + c;
    float* q = agg + (size_t)t * d + c;
    unsigned long long v = *(const unsigned long long*)p;
    atomicAdd(q + 0, bf2f((unsigned short)(v)));
    atomicAdd(q + 1, bf2f((unsigned short)(v >> 16)));
    atomicAdd(q + 2, bf2f((unsigned short)(v >> 32)));
    atomicAdd(q + 3, bf2f((unsigned short)(v >> 48)));
}

// ---------------- mean finalize: agg_bf16 = agg_f32 / max(cnt,1) ----------
__global__ void finalize_kernel(const float* __restrict__ agg,
                                const float* __restrict__ cnt,
                                unsigned short* __restrict__ out,
                                int total, int log2d) {
    int i = blockIdx.x * blockDim.x + threadIdx.x;
    if (i >= total) return;
    float c = fmaxf(cnt[i >> log2d], 1.0f);
    out[i] = f2bf(agg[i] / c);
}

// ---------------- fused SAGE GEMM:  out = act(A1@B1t + A2@B2t + bias) ------
// A1/A2: [M,K] bf16 row-major.  B1/B2: [NOUT,K] bf16 (N-major == B^T).
// One wave owns one 16-row M-tile and ALL N-tiles: A fragments loaded once
// per K-step and reused across NOUT/16 accumulators (kills redundant A
// traffic).  B1/B2 staged in LDS once per block (padded rows, b128 reads).
union Frag { v16bf v; ushort8 u[2]; };

template <int K, int NOUT>
__global__ __launch_bounds__(256)
void sage_gemm_kernel(const unsigned short* __restrict__ A1,
                      const unsigned short* __restrict__ A2,
                      const unsigned short* __restrict__ B1,
                      const unsigned short* __restrict__ B2,
                      const float* __restrict__ bias,
                      unsigned short* __restrict__ out,
                      int M, int relu) {
    constexpr int NT = NOUT / 16;          // N tiles per wave
    constexpr int RS = K + 8;              // LDS row stride (halfs): 272B, 16B-aligned
    extern __shared__ unsigned short smem[];
    unsigned short* sB1 = smem;
    unsigned short* sB2 = smem + (size_t)NOUT * RS;

    // ---- cooperative B1/B2 stage into LDS (16B chunks) ----
    constexpr int CH = K / 8;              // chunks per row
    for (int idx = threadIdx.x; idx < NOUT * CH; idx += 256) {
        int r = idx / CH, c = (idx - r * CH) * 8;
        *(ushort8*)(sB1 + r * RS + c) = *(const ushort8*)(B1 + (size_t)r * K + c);
        *(ushort8*)(sB2 + r * RS + c) = *(const ushort8*)(B2 + (size_t)r * K + c);
    }
    __syncthreads();

    int wave = blockIdx.x * 8 + (threadIdx.x >> 5);   // = mtile
    int lane = threadIdx.x & 31;
    if (wave * 16 >= M) return;                       // wave-uniform, EXEC stays ~0

    int half16 = (lane >> 4) * 8;                     // K-half this lane feeds
    int lrow = lane & 15;
    const unsigned short* a1p = A1 + ((size_t)(wave * 16 + lrow)) * K + half16;
    const unsigned short* a2p = A2 + ((size_t)(wave * 16 + lrow)) * K + half16;

    v8f acc[NT] = {};
#pragma unroll
    for (int kk = 0; kk < K; kk += 32) {
        Frag a1, a2;
        // ISA layout: VGPR0-3 = 8 contiguous K at 8*(lane>>4); VGPR4-7 = +16
        a1.u[0] = *(const ushort8*)(a1p + kk);
        a1.u[1] = *(const ushort8*)(a1p + kk + 16);
        a2.u[0] = *(const ushort8*)(a2p + kk);
        a2.u[1] = *(const ushort8*)(a2p + kk + 16);
#pragma unroll
        for (int nt = 0; nt < NT; ++nt) {
            const unsigned short* bb = sB1 + (size_t)(nt * 16 + lrow) * RS + kk + half16;
            Frag b;
            b.u[0] = *(const ushort8*)(bb);
            b.u[1] = *(const ushort8*)(bb + 16);
            acc[nt] = __builtin_amdgcn_wmma_f32_16x16x32_bf16(
                false, a1.v, false, b.v, (short)0, acc[nt], false, false);
            bb = sB2 + (size_t)(nt * 16 + lrow) * RS + kk + half16;
            b.u[0] = *(const ushort8*)(bb);
            b.u[1] = *(const ushort8*)(bb + 16);
            acc[nt] = __builtin_amdgcn_wmma_f32_16x16x32_bf16(
                false, a2.v, false, b.v, (short)0, acc[nt], false, false);
        }
    }

    // C/D layout: lane&15 = N col; VGPR r -> M row r + 8*(lane>>4)
    int rbase = wave * 16 + half16;
#pragma unroll
    for (int nt = 0; nt < NT; ++nt) {
        int col = nt * 16 + lrow;
        float bv = bias[col];
#pragma unroll
        for (int r = 0; r < 8; ++r) {
            float v = acc[nt][r] + bv;
            if (relu) v = fmaxf(v, 0.0f);
            out[(size_t)(rbase + r) * NOUT + col] = f2bf(v);
        }
    }
}

// ---------------- final layer: 2 logits + log_softmax (fp32 out) ----------
__global__ void sage_out_kernel(const unsigned short* __restrict__ agg,
                                const unsigned short* __restrict__ h,
                                const float* __restrict__ Wl,
                                const float* __restrict__ Wr,
                                const float* __restrict__ b3,
                                float* __restrict__ out, int n, int K) {
    int i = blockIdx.x * blockDim.x + threadIdx.x;
    if (i >= n) return;
    const unsigned short* ap = agg + (size_t)i * K;
    const unsigned short* hp = h + (size_t)i * K;
    float z0 = b3[0], z1 = b3[1];
    for (int k = 0; k < K; ++k) {
        float a = bf2f(ap[k]);
        float x = bf2f(hp[k]);
        z0 += a * Wl[2 * k + 0] + x * Wr[2 * k + 0];
        z1 += a * Wl[2 * k + 1] + x * Wr[2 * k + 1];
    }
    float m = fmaxf(z0, z1);
    float lse = m + __logf(__expf(z0 - m) + __expf(z1 - m));
    out[2 * i + 0] = z0 - lse;
    out[2 * i + 1] = z1 - lse;
}

// ======================================================================
extern "C" void kernel_launch(void* const* d_in, const int* in_sizes, int n_in,
                              void* d_out, int out_size, void* d_ws, size_t ws_size,
                              hipStream_t stream) {
    const float* x   = (const float*)d_in[0];
    const int*   ei  = (const int*)d_in[1];
    const float* W1l = (const float*)d_in[2];
    const float* W1r = (const float*)d_in[3];
    const float* b1  = (const float*)d_in[4];
    const float* W2l = (const float*)d_in[5];
    const float* W2r = (const float*)d_in[6];
    const float* b2  = (const float*)d_in[7];
    const float* W3l = (const float*)d_in[8];
    const float* W3r = (const float*)d_in[9];
    const float* b3  = (const float*)d_in[10];
    float* out = (float*)d_out;

    const int P = 128, HID = 128, H2 = 256;
    const int nNodes = in_sizes[0] / P;
    const int nEdges = in_sizes[1] / 2;
    const int* srcI = ei;
    const int* dstI = ei + nEdges;

    // ---- workspace carve-up ----
    char* ws = (char*)d_ws;
    size_t off = 0;
    auto carve = [&](size_t bytes) -> void* {
        void* p = ws + off;
        off = (off + bytes + 255) & ~(size_t)255;
        return p;
    };
    unsigned short* xbf   = (unsigned short*)carve((size_t)nNodes * P   * 2);
    unsigned short* h1bf  = (unsigned short*)carve((size_t)nNodes * HID * 2);
    unsigned short* h2bf  = (unsigned short*)carve((size_t)nNodes * H2  * 2);
    float*          aggf  = (float*)        carve((size_t)nNodes * H2  * 4);
    unsigned short* aggbf = (unsigned short*)carve((size_t)nNodes * H2  * 2);
    float*          cnt   = (float*)        carve((size_t)nNodes * 4);
    unsigned short* w1lt  = (unsigned short*)carve((size_t)P * HID * 2);
    unsigned short* w1rt  = (unsigned short*)carve((size_t)P * HID * 2);
    unsigned short* w2lt  = (unsigned short*)carve((size_t)HID * H2 * 2);
    unsigned short* w2rt  = (unsigned short*)carve((size_t)HID * H2 * 2);

    const int T = 256;
    auto blks = [](long long total, int t) { return (unsigned)((total + t - 1) / t); };

    // ---- degree counts + static conversions ----
    hipMemsetAsync(cnt, 0, (size_t)nNodes * 4, stream);
    count_kernel<<<blks(nEdges, T), T, 0, stream>>>(dstI, cnt, nEdges);
    cvt_bf16_kernel<<<blks((long long)nNodes * P, T), T, 0, stream>>>(x, xbf, nNodes * P);
    wt_kernel<<<blks(P * HID, T), T, 0, stream>>>(W1l, w1lt, P, HID);
    wt_kernel<<<blks(P * HID, T), T, 0, stream>>>(W1r, w1rt, P, HID);
    wt_kernel<<<blks(HID * H2, T), T, 0, stream>>>(W2l, w2lt, HID, H2);
    wt_kernel<<<blks(HID * H2, T), T, 0, stream>>>(W2r, w2rt, HID, H2);

    const int mtiles = (nNodes + 15) / 16;
    const unsigned gemmBlocks = (unsigned)((mtiles + 7) / 8);
    const size_t lds1 = (size_t)2 * HID * (128 + 8) * 2;   // 69,632 B
    const size_t lds2 = (size_t)2 * H2  * (128 + 8) * 2;   // 139,264 B

    // ---- layer 1: agg(x) -> h1 = relu(agg@W1l + x@W1r + b1)  [N,128] ----
    hipMemsetAsync(aggf, 0, (size_t)nNodes * P * 4, stream);
    scatter_kernel<<<blks((long long)nEdges * (P / 4), T), T, 0, stream>>>(
        xbf, srcI, dstI, aggf, nEdges, P, 5);
    finalize_kernel<<<blks((long long)nNodes * P, T), T, 0, stream>>>(
        aggf, cnt, aggbf, nNodes * P, 7);
    sage_gemm_kernel<128, 128><<<gemmBlocks, T, lds1, stream>>>(
        aggbf, xbf, w1lt, w1rt, b1, h1bf, nNodes, 1);

    // ---- layer 2: agg(h1) -> h2 = relu(agg@W2l + h1@W2r + b2) [N,256] ----
    hipMemsetAsync(aggf, 0, (size_t)nNodes * HID * 4, stream);
    scatter_kernel<<<blks((long long)nEdges * (HID / 4), T), T, 0, stream>>>(
        h1bf, srcI, dstI, aggf, nEdges, HID, 5);
    finalize_kernel<<<blks((long long)nNodes * HID, T), T, 0, stream>>>(
        aggf, cnt, aggbf, nNodes * HID, 7);
    sage_gemm_kernel<128, 256><<<gemmBlocks, T, lds2, stream>>>(
        aggbf, h1bf, w2lt, w2rt, b2, h2bf, nNodes, 1);

    // ---- layer 3: agg(h2) -> logits -> log_softmax [N,2] fp32 ----
    hipMemsetAsync(aggf, 0, (size_t)nNodes * H2 * 4, stream);
    scatter_kernel<<<blks((long long)nEdges * (H2 / 4), T), T, 0, stream>>>(
        h2bf, srcI, dstI, aggf, nEdges, H2, 6);
    finalize_kernel<<<blks((long long)nNodes * H2, T), T, 0, stream>>>(
        aggf, cnt, aggbf, nNodes * H2, 8);
    sage_out_kernel<<<blks(nNodes, T), T, 0, stream>>>(
        aggbf, h2bf, W3l, W3r, b3, out, nNodes, H2);
}
